// InfiniTransformer_42855183680151
// MI455X (gfx1250) — matrix operand order repaired and checked by reading, not compiled, not run on hardware
//
#include <hip/hip_runtime.h>
#include <hip/hip_bf16.h>
#include <math.h>

typedef _Float16 half_t;
typedef _Float16 v16h __attribute__((ext_vector_type(16)));
typedef _Float16 v8h  __attribute__((ext_vector_type(8)));
typedef float    v8f  __attribute__((ext_vector_type(8)));
typedef float    v4f  __attribute__((ext_vector_type(4)));
typedef unsigned int u32x4 __attribute__((ext_vector_type(4)));
typedef int          i32x8 __attribute__((ext_vector_type(8)));
typedef int          i32x4 __attribute__((ext_vector_type(4)));

#define B_    2
#define S_    8192
#define D_    1024
#define DH_   4096
#define H_    8
#define DK_   128
#define DV_   128
#define L_    512
#define NSEG_ 16
#define M_    (B_*S_)   /* 16384 */

// ---------------------------------------------------------------- helpers
__device__ inline v8f zero8() {
  v8f r;
#pragma unroll
  for (int t = 0; t < 8; t++) r[t] = 0.f;
  return r;
}
__device__ inline v16h mk16(v8h lo, v8h hi) {
  v16h r;
#pragma unroll
  for (int t = 0; t < 8; t++) { r[t] = lo[t]; r[t + 8] = hi[t]; }
  return r;
}
// A fragment (16x32 f16, one 16-row tile): lane L holds row M=L%16.
// Half-wave 0 holds K = {klo..klo+7, 16+klo..}, klo = (lane>=16)*8.
__device__ inline v16h ldfa(const half_t* rowp, int klo) {
  v8h lo = *(const v8h*)(rowp + klo);
  v8h hi = *(const v8h*)(rowp + 16 + klo);
  return mk16(lo, hi);
}
// B fragment (32x16 f16): lane L holds column N=L%16; K = {klo..klo+7, 16+klo..}, same as A.
__device__ inline v16h ldfb(const half_t* p, int klo, int ld) {
  v16h r;
#pragma unroll
  for (int t = 0; t < 8; t++) { r[t] = p[(klo + t) * ld]; r[t + 8] = p[(16 + klo + t) * ld]; }
  return r;
}
__device__ inline v8f wmma_f16(v16h a, v16h b, v8f c) {
  return __builtin_amdgcn_wmma_f32_16x16x32_f16(
      /*neg_a=*/false, a, /*neg_b=*/false, b,
      /*c_mod=*/(short)0, c, /*reuse_a=*/false, /*reuse_b=*/false);
}

// ---------------------------------------------------------------- TDM 2D tile load (f16 elements)
// Builds a D# per CDNA5 ISA and issues TENSOR_LOAD_TO_LDS.
// pad_interval/pad_amount reproduce the bank-padded LDS strides (codes per ISA:
// interval 3 -> every 16 DWORDs, amount 3 -> 4 DWORDs; interval 5 -> every 64 DWORDs).
// This toolchain exposes the 6-arg builtin: (g0, g1, g2, g3, g_extra, cpol).
__device__ inline void tdm_load_2d_f16(unsigned lds_off, const void* gptr,
                                       unsigned tile_w, unsigned tile_h,
                                       unsigned tensor_w, unsigned tensor_h,
                                       unsigned row_stride,
                                       unsigned pad_interval, unsigned pad_amount) {
  unsigned long long ga = (unsigned long long)(uintptr_t)gptr;
  u32x4 g0;
  g0[0] = 1u;                                            // count=1, user descriptor
  g0[1] = lds_off;                                       // lds_addr (bytes)
  g0[2] = (unsigned)(ga & 0xffffffffu);                  // global_addr[31:0]
  g0[3] = (unsigned)((ga >> 32) & 0x01ffffffu) | (2u << 30);  // addr[56:32] | type=2
  i32x8 g1;
  unsigned pe = (pad_amount | pad_interval) ? 1u : 0u;
  g1[0] = (int)((1u << 16) | (pe << 20) | (pad_interval << 22) | (pad_amount << 25)); // data_size=2B
  g1[1] = (int)((tensor_w & 0xffffu) << 16);                               // tensor_dim0[15:0]
  g1[2] = (int)(((tensor_w >> 16) & 0xffffu) | ((tensor_h & 0xffffu) << 16)); // dim0 hi | dim1 lo
  g1[3] = (int)(((tensor_h >> 16) & 0xffffu) | ((tile_w & 0xffffu) << 16));   // dim1 hi | tile_dim0
  g1[4] = (int)(tile_h & 0xffffu);                                         // tile_dim1 (tile_dim2=0)
  g1[5] = (int)row_stride;                                                 // tensor_dim0_stride lo
  g1[6] = 0;                                                               // stride hi | dim1_stride lo
  g1[7] = 0;
  i32x4 zz;
  zz[0] = 0; zz[1] = 0; zz[2] = 0; zz[3] = 0;
  i32x8 z8;
#pragma unroll
  for (int t = 0; t < 8; t++) z8[t] = 0;
  __builtin_amdgcn_tensor_load_to_lds(g0, g1, zz, zz, z8, 0);
}

// ---------------------------------------------------------------- fp32 -> f16
__global__ __launch_bounds__(256) void cvt_f2h(const float* __restrict__ in,
                                               half_t* __restrict__ out,
                                               long long n) {
  long long i = (long long)blockIdx.x * blockDim.x + threadIdx.x;
  long long stride = (long long)gridDim.x * blockDim.x;
  for (; i < n; i += stride) out[i] = (half_t)in[i];
}

// ---------------------------------------------------------------- generic batched WMMA GEMM
// C(M,N) = epi( A(M,K) * B^T + ... ),  A f16 (or f32 if AF32), B f16.
// BLAYOUT 0: B stored (N,K) row-major (weights).  BLAYOUT 1: B stored (K,N).
// EPI 0: f16 store.  1: f32 store *alpha.  2: +bias, relu, f16.  3: +bias +resid, f32.
template <int BLAYOUT, int AF32, int EPI>
__global__ __launch_bounds__(256)
void gemm_wmma_k(const void* __restrict__ Aptr, const half_t* __restrict__ Bptr,
                 const float* __restrict__ bias, const float* __restrict__ resid,
                 float* __restrict__ outF, half_t* __restrict__ outH,
                 int M, int N, int K, int lda, int ldb, int ldc, int ldres,
                 int Hn, int NSeg,
                 long long sAb, long long sAh, long long sAs,
                 long long sBb, long long sBh, long long sBs,
                 long long sCb, long long sCh, long long sCs,
                 float alpha) {
  __shared__ half_t As[128 * 40];   // 128 rows x 32 K (+4dw pad / 16dw via TDM)
  __shared__ half_t Bs[32 * 136];   // 32 K x 128 N (+4dw pad / 64dw), K-major

  int z = blockIdx.z;
  int zs = z % NSeg; int t = z / NSeg; int zh = t % Hn; int zb = t / Hn;
  long long offA = (long long)zb * sAb + (long long)zh * sAh + (long long)zs * sAs;
  long long offB = (long long)zb * sBb + (long long)zh * sBh + (long long)zs * sBs;
  long long offC = (long long)zb * sCb + (long long)zh * sCh + (long long)zs * sCs;

  const half_t* Ah = (const half_t*)Aptr + offA;
  const float*  Af = (const float*)Aptr + offA;
  const half_t* Bp = Bptr + offB;

  int m0 = blockIdx.y * 128, n0 = blockIdx.x * 128;
  int tid = threadIdx.x, lane = tid & 31, wave = tid >> 5;
  int wm = (wave >> 2) * 64, wn = (wave & 3) * 32;
  int rlo = (lane >> 4) * 8, cn = lane & 15, klo = rlo;

  unsigned asoff = (unsigned)(unsigned long long)(uintptr_t)(void*)&As[0];
  unsigned bsoff = (unsigned)(unsigned long long)(uintptr_t)(void*)&Bs[0];

  v8f acc[4][2];
#pragma unroll
  for (int i = 0; i < 4; i++) { acc[i][0] = zero8(); acc[i][1] = zero8(); }

  for (int k0 = 0; k0 < K; k0 += 32) {
    // ---- stage A tile (128 x 32)
    if (AF32) {
#pragma unroll
      for (int it = 0; it < 2; ++it) {
        int idx = tid + it * 256;          // 512 v8h chunks
        int row = idx >> 2, c8 = (idx & 3) << 3;
        const float* pf = Af + (long long)(m0 + row) * lda + k0 + c8;
        v4f a0 = *(const v4f*)pf, a1 = *(const v4f*)(pf + 4);
        v8h hv;
#pragma unroll
        for (int j = 0; j < 4; j++) { hv[j] = (half_t)a0[j]; hv[4 + j] = (half_t)a1[j]; }
        *(v8h*)&As[row * 40 + c8] = hv;
      }
    } else {
      // Tensor Data Mover: one DMA for the whole 128x32 f16 tile, padded to
      // a 40-half LDS row stride (pad 4 DWORDs after every 16 DWORDs).
      if (wave == 0)
        tdm_load_2d_f16(asoff, Ah + (long long)m0 * lda + k0,
                        32u, 128u, (unsigned)(K - k0), (unsigned)(M - m0),
                        (unsigned)lda, 3u, 3u);
    }
    // ---- stage B tile, K-major [32][128+pad]
    if (BLAYOUT == 0) {
#pragma unroll
      for (int it = 0; it < 2; ++it) {
        int idx = tid + it * 256;
        int n = idx >> 2, c8 = (idx & 3) << 3;
        v8h bv = *(const v8h*)(Bp + (long long)(n0 + n) * ldb + k0 + c8);
#pragma unroll
        for (int j = 0; j < 8; j++) Bs[(c8 + j) * 136 + n] = bv[j];
        if (k0 + 32 < K)
          __builtin_prefetch(Bp + (long long)(n0 + n) * ldb + k0 + 32 + c8, 0, 1);
      }
    } else {
      // (K,N) layout: rectangular tile -> TDM, pad 4 DWORDs after every 64 DWORDs
      if (wave == 0)
        tdm_load_2d_f16(bsoff, Bp + (long long)k0 * ldb + n0,
                        128u, 32u, (unsigned)(N - n0), (unsigned)(K - k0),
                        (unsigned)ldb, 5u, 3u);
    }
    if ((!AF32 || BLAYOUT == 1) && wave == 0)
      __builtin_amdgcn_s_wait_tensorcnt(0);
    __syncthreads();

    v16h bf0 = ldfb(&Bs[wn + cn], klo, 136);
    v16h bf1 = ldfb(&Bs[wn + 16 + cn], klo, 136);
#pragma unroll
    for (int i = 0; i < 4; i++) {
      v16h af = ldfa(&As[(wm + i * 16 + cn) * 40], klo);
      acc[i][0] = wmma_f16(af, bf0, acc[i][0]);
      acc[i][1] = wmma_f16(af, bf1, acc[i][1]);
    }
    __syncthreads();
  }

  // ---- epilogue: C fragment f32 16x16 -> M = r + 8*(lane>=16), N = lane%16
#pragma unroll
  for (int i = 0; i < 4; i++)
#pragma unroll
    for (int j = 0; j < 2; j++)
#pragma unroll
      for (int r = 0; r < 8; r++) {
        int gm = m0 + wm + i * 16 + rlo + r;
        int gn = n0 + wn + j * 16 + cn;
        long long ci = offC + (long long)gm * ldc + gn;
        float v = acc[i][j][r];
        if (EPI == 0) {
          outH[ci] = (half_t)v;
        } else if (EPI == 1) {
          outF[ci] = v * alpha;
        } else if (EPI == 2) {
          float t2 = v + bias[gn];
          outH[ci] = (half_t)(t2 > 0.f ? t2 : 0.f);
        } else {
          outF[ci] = v + bias[gn] + resid[(long long)gm * ldres + gn];
        }
      }
}

// ---------------------------------------------------------------- softmax over 512-wide rows, in place
__global__ __launch_bounds__(256) void softmax512_k(float* __restrict__ sc) {
  __shared__ float red[256];
  long long row = blockIdx.x;
  float* p = sc + row * 512;
  int tid = threadIdx.x;
  float a = p[tid], b = p[tid + 256];
  red[tid] = fmaxf(a, b);
  __syncthreads();
  for (int s = 128; s > 0; s >>= 1) {
    if (tid < s) red[tid] = fmaxf(red[tid], red[tid + s]);
    __syncthreads();
  }
  float mx = red[0];
  __syncthreads();
  float e0 = __expf(a - mx), e1 = __expf(b - mx);
  red[tid] = e0 + e1;
  __syncthreads();
  for (int s = 128; s > 0; s >>= 1) {
    if (tid < s) red[tid] += red[tid + s];
    __syncthreads();
  }
  float inv = 1.0f / red[0];
  p[tid] = e0 * inv;
  p[tid + 256] = e1 * inv;
}

// ---------------------------------------------------------------- Infini-memory sequential scan
// One workgroup per (b,h); mem (128x128 f32) lives in LDS; WMMA for both
// sq@mem retrieval and sk^T@v update; z/denominator via LDS float atomics;
// V tiles fetched by the Tensor Data Mover.
__global__ __launch_bounds__(256)
void mem_scan_k(const half_t* __restrict__ qh, const half_t* __restrict__ kh,
                const half_t* __restrict__ vh, const float* __restrict__ attdot,
                const float* __restrict__ betas, half_t* __restrict__ atth) {
  extern __shared__ char smem[];
  float* memf  = (float*)smem;           // 128*128
  float* zf    = memf + 128 * 128;       // 128
  float* denom = zf + 128;               // 128
  float* betaS = denom + 128;            // 128
  half_t* As   = (half_t*)(betaS + 128); // 128*136
  half_t* Bs   = As + 128 * 136;         // 32*136

  int tid = threadIdx.x, lane = tid & 31, wave = tid >> 5;
  int b = blockIdx.x >> 3, h = blockIdx.x & 7;
  int wm = (wave >> 2) * 64, wn = (wave & 3) * 32;
  int rlo = (lane >> 4) * 8, cn = lane & 15, klo = rlo;
  const long long RS = 1024;  // H*dk row stride
  unsigned bsoff = (unsigned)(unsigned long long)(uintptr_t)(void*)Bs;

  for (int i = tid; i < 128 * 128; i += 256) memf[i] = 0.f;
  if (tid < 128) {
    zf[tid] = 1.0f / 128.0f;
    betaS[tid] = 1.0f / (1.0f + __expf(-betas[h * 128 + tid]));
  }
  __syncthreads();

  for (int seg = 0; seg < NSEG_; ++seg) {
    long long srow0 = (long long)b * S_ + (long long)seg * L_;

    // ===== Phase 1: att_mem = sq@mem / sq@z, combine with att_dot (4 chunks of 128 rows)
    for (int c = 0; c < 4; ++c) {
      int i0 = c * 128;
      if (tid < 128) denom[tid] = 0.f;
      __syncthreads();
      // stage sq (elu(q)+1) into As[128][128] and accumulate denom
      for (int idx = tid; idx < 2048; idx += 256) {
        int row = idx >> 4, c8 = (idx & 15) << 3;
        v8h qv = *(const v8h*)(qh + (srow0 + i0 + row) * RS + h * 128 + c8);
        float part = 0.f;
#pragma unroll
        for (int j = 0; j < 8; j++) {
          float x = (float)qv[j];
          float s = x > 0.f ? x + 1.f : __expf(x);
          part += s * zf[c8 + j];
          As[row * 136 + c8 + j] = (half_t)s;
        }
        atomicAdd(&denom[row], part);
      }
      __syncthreads();
      // one v_rcp_f32 per row instead of 64 IEEE divides per lane later
      if (tid < 128) denom[tid] = __builtin_amdgcn_rcpf(denom[tid]);

      v8f acc[4][2];
#pragma unroll
      for (int i = 0; i < 4; i++) { acc[i][0] = zero8(); acc[i][1] = zero8(); }
      for (int k0 = 0; k0 < 128; k0 += 32) {
        // stage mem chunk as f16 B tile (K-major)
        for (int idx = tid; idx < 512; idx += 256) {
          int kk = idx >> 4, n8 = (idx & 15) << 3;
#pragma unroll
          for (int j = 0; j < 8; j++)
            Bs[kk * 136 + n8 + j] = (half_t)memf[(k0 + kk) * 128 + n8 + j];
        }
        __syncthreads();
        v16h bf0 = ldfb(&Bs[wn + cn], klo, 136);
        v16h bf1 = ldfb(&Bs[wn + 16 + cn], klo, 136);
#pragma unroll
        for (int i = 0; i < 4; i++) {
          v16h af = ldfa(&As[(wm + i * 16 + cn) * 136 + k0], klo);
          acc[i][0] = wmma_f16(af, bf0, acc[i][0]);
          acc[i][1] = wmma_f16(af, bf1, acc[i][1]);
        }
        __syncthreads();
      }
      // combine: att = beta*att_mem + (1-beta)*att_dot
#pragma unroll
      for (int i = 0; i < 4; i++)
#pragma unroll
        for (int j = 0; j < 2; j++)
#pragma unroll
          for (int r = 0; r < 8; r++) {
            int li = wm + i * 16 + rlo + r;
            int e = wn + j * 16 + cn;
            long long srow = srow0 + i0 + li;
            float am = acc[i][j][r] * denom[li];
            float bt = betaS[e];
            long long oidx = srow * RS + h * 128 + e;
            float ad = attdot[oidx];
            atth[oidx] = (half_t)(bt * am + (1.f - bt) * ad);
          }
      __syncthreads();
    }

    // ===== Phase 2: mem += sk^T @ v ; z += sum_i sk
    v8f macc[4][2];
#pragma unroll
    for (int i = 0; i < 4; i++) { macc[i][0] = zero8(); macc[i][1] = zero8(); }
    for (int kc = 0; kc < 16; ++kc) {
      int i0 = kc * 32;
      // V tile (32 x 128 f16, contiguous rows) via TDM, padded to 136-half rows
      if (wave == 0)
        tdm_load_2d_f16(bsoff, vh + (srow0 + i0) * RS + h * 128,
                        128u, 32u, 128u, 32u, (unsigned)RS, 5u, 3u);
      for (int idx = tid; idx < 512; idx += 256) {
        int ii = idx >> 4, d8 = (idx & 15) << 3;
        v8h kv = *(const v8h*)(kh + (srow0 + i0 + ii) * RS + h * 128 + d8);
#pragma unroll
        for (int j = 0; j < 8; j++) {
          float x = (float)kv[j];
          float s = x > 0.f ? x + 1.f : __expf(x);
          As[(d8 + j) * 136 + ii] = (half_t)s;   // sk^T (d-major)
          atomicAdd(&zf[d8 + j], s);
        }
      }
      if (wave == 0) __builtin_amdgcn_s_wait_tensorcnt(0);
      __syncthreads();
      v16h bf0 = ldfb(&Bs[wn + cn], klo, 136);
      v16h bf1 = ldfb(&Bs[wn + 16 + cn], klo, 136);
#pragma unroll
      for (int i = 0; i < 4; i++) {
        v16h af = ldfa(&As[(wm + i * 16 + cn) * 136], klo);
        macc[i][0] = wmma_f16(af, bf0, macc[i][0]);
        macc[i][1] = wmma_f16(af, bf1, macc[i][1]);
      }
      __syncthreads();
    }
#pragma unroll
    for (int i = 0; i < 4; i++)
#pragma unroll
      for (int j = 0; j < 2; j++)
#pragma unroll
        for (int r = 0; r < 8; r++) {
          int d = wm + i * 16 + rlo + r;
          int e = wn + j * 16 + cn;
          memf[d * 128 + e] += macc[i][j][r];
        }
    __syncthreads();
  }
}

// ---------------------------------------------------------------- layernorm over D=1024 rows
__global__ __launch_bounds__(256)
void layernorm_k(const float* __restrict__ y, const float* __restrict__ g,
                 const float* __restrict__ bta, float* __restrict__ out) {
  __shared__ float red[256];
  long long row = blockIdx.x;
  const float* p = y + row * 1024;
  int tid = threadIdx.x;
  float s = 0.f;
  for (int i = tid; i < 1024; i += 256) s += p[i];
  red[tid] = s;
  __syncthreads();
  for (int st = 128; st > 0; st >>= 1) {
    if (tid < st) red[tid] += red[tid + st];
    __syncthreads();
  }
  float mu = red[0] * (1.0f / 1024.0f);
  __syncthreads();
  float v = 0.f;
  for (int i = tid; i < 1024; i += 256) { float d = p[i] - mu; v += d * d; }
  red[tid] = v;
  __syncthreads();
  for (int st = 128; st > 0; st >>= 1) {
    if (tid < st) red[tid] += red[tid + st];
    __syncthreads();
  }
  float inv = rsqrtf(red[0] * (1.0f / 1024.0f) + 1e-5f);
  for (int i = tid; i < 1024; i += 256)
    out[row * 1024 + i] = (p[i] - mu) * inv * g[i] + bta[i];
}

// ---------------------------------------------------------------- host
extern "C" void kernel_launch(void* const* d_in, const int* in_sizes, int n_in,
                              void* d_out, int out_size, void* d_ws, size_t ws_size,
                              hipStream_t stream) {
  (void)in_sizes; (void)n_in; (void)out_size; (void)ws_size;
  const float* x    = (const float*)d_in[0];
  const float* Wq   = (const float*)d_in[1];
  const float* Wk   = (const float*)d_in[2];
  const float* Wv   = (const float*)d_in[3];
  const float* Wo   = (const float*)d_in[4];
  const float* bet  = (const float*)d_in[5];
  const float* W1   = (const float*)d_in[6];
  const float* b1   = (const float*)d_in[7];
  const float* W2   = (const float*)d_in[8];
  const float* b2   = (const float*)d_in[9];
  const float* ln_g = (const float*)d_in[10];
  const float* ln_b = (const float*)d_in[11];

  char* ws = (char*)d_ws;
  const size_t MB = 1024ull * 1024ull;
  half_t* WQH = (half_t*)(ws + 0 * MB);
  half_t* WKH = (half_t*)(ws + 2 * MB);
  half_t* WVH = (half_t*)(ws + 4 * MB);
  half_t* WOH = (half_t*)(ws + 6 * MB);
  half_t* W1H = (half_t*)(ws + 8 * MB);
  half_t* W2H = (half_t*)(ws + 16 * MB);
  half_t* XH  = (half_t*)(ws + 24 * MB);
  half_t* QH  = (half_t*)(ws + 56 * MB);
  half_t* KH  = (half_t*)(ws + 88 * MB);
  half_t* VH  = (half_t*)(ws + 120 * MB);
  float*  SC  = (float*)(ws + 152 * MB);   // scores / probs (256 MB, in place)
  half_t* ATTH= (half_t*)(ws + 152 * MB);  // reuse after probs consumed
  half_t* AOH = (half_t*)(ws + 184 * MB);
  half_t* H1H = (half_t*)(ws + 216 * MB);
  float*  AD  = (float*)(ws + 408 * MB);   // att_dot (64 MB)
  float*  YF  = (float*)(ws + 56 * MB);    // reuse Q/K region after scan

  const long long Z = 0LL;
  const float scale = 0.0883883476483184f;  // 1/sqrt(128)

  // ---- fp32 -> f16 conversions
  cvt_f2h<<<2048, 256, 0, stream>>>(x,  XH,  (long long)M_ * D_);
  cvt_f2h<<<1024, 256, 0, stream>>>(Wq, WQH, (long long)D_ * D_);
  cvt_f2h<<<1024, 256, 0, stream>>>(Wk, WKH, (long long)D_ * D_);
  cvt_f2h<<<1024, 256, 0, stream>>>(Wv, WVH, (long long)D_ * D_);
  cvt_f2h<<<1024, 256, 0, stream>>>(Wo, WOH, (long long)D_ * D_);
  cvt_f2h<<<2048, 256, 0, stream>>>(W1, W1H, (long long)DH_ * D_);
  cvt_f2h<<<2048, 256, 0, stream>>>(W2, W2H, (long long)D_ * DH_);

  // ---- QKV projections: (16384 x 1024) = XH @ W^T
  dim3 gq(D_ / 128, M_ / 128, 1);
  gemm_wmma_k<0, 0, 0><<<gq, 256, 0, stream>>>(XH, WQH, nullptr, nullptr, nullptr, QH,
      M_, D_, D_, D_, D_, D_, 0, 1, 1, Z,Z,Z, Z,Z,Z, Z,Z,Z, 1.0f);
  gemm_wmma_k<0, 0, 0><<<gq, 256, 0, stream>>>(XH, WKH, nullptr, nullptr, nullptr, KH,
      M_, D_, D_, D_, D_, D_, 0, 1, 1, Z,Z,Z, Z,Z,Z, Z,Z,Z, 1.0f);
  gemm_wmma_k<0, 0, 0><<<gq, 256, 0, stream>>>(XH, WVH, nullptr, nullptr, nullptr, VH,
      M_, D_, D_, D_, D_, D_, 0, 1, 1, Z,Z,Z, Z,Z,Z, Z,Z,Z, 1.0f);

  // ---- local attention scores per (b,h,seg): S = q k^T * scale
  dim3 gs(L_ / 128, L_ / 128, B_ * H_ * NSEG_);
  gemm_wmma_k<0, 0, 1><<<gs, 256, 0, stream>>>(QH, KH, nullptr, nullptr, SC, nullptr,
      L_, L_, DK_, 1024, 1024, L_, 0, H_, NSEG_,
      (long long)S_ * 1024, 128LL, (long long)L_ * 1024,
      (long long)S_ * 1024, 128LL, (long long)L_ * 1024,
      (long long)H_ * NSEG_ * L_ * L_, (long long)NSEG_ * L_ * L_, (long long)L_ * L_,
      scale);

  // ---- softmax (in place)
  softmax512_k<<<B_ * H_ * NSEG_ * L_, 256, 0, stream>>>(SC);

  // ---- att_dot = P @ V   (P is fp32, cast to f16 at LDS staging; V tile via TDM)
  dim3 ga(DV_ / 128, L_ / 128, B_ * H_ * NSEG_);
  gemm_wmma_k<1, 1, 1><<<ga, 256, 0, stream>>>(SC, VH, nullptr, nullptr, AD, nullptr,
      L_, DV_, L_, L_, 1024, 1024, 0, H_, NSEG_,
      (long long)H_ * NSEG_ * L_ * L_, (long long)NSEG_ * L_ * L_, (long long)L_ * L_,
      (long long)S_ * 1024, 128LL, (long long)L_ * 1024,
      (long long)S_ * 1024, 128LL, (long long)L_ * 1024,
      1.0f);

  // ---- sequential memory scan + beta-gated combine
  const int scan_lds = (128 * 128 + 128 + 128 + 128) * 4 + (128 * 136 + 32 * 136) * 2;
  (void)hipFuncSetAttribute(reinterpret_cast<const void*>(mem_scan_k),
                            hipFuncAttributeMaxDynamicSharedMemorySize, scan_lds);
  mem_scan_k<<<B_ * H_, 256, scan_lds, stream>>>(QH, KH, VH, AD, bet, ATTH);

  // ---- output projection
  gemm_wmma_k<0, 0, 0><<<gq, 256, 0, stream>>>(ATTH, WOH, nullptr, nullptr, nullptr, AOH,
      M_, D_, H_ * DV_, 1024, 1024, D_, 0, 1, 1, Z,Z,Z, Z,Z,Z, Z,Z,Z, 1.0f);

  // ---- MLP: relu(ao @ W1^T + b1)
  dim3 g1(DH_ / 128, M_ / 128, 1);
  gemm_wmma_k<0, 0, 2><<<g1, 256, 0, stream>>>(AOH, W1H, b1, nullptr, nullptr, H1H,
      M_, DH_, D_, D_, D_, DH_, 0, 1, 1, Z,Z,Z, Z,Z,Z, Z,Z,Z, 1.0f);

  // ---- MLP out + bias + residual x  (fp32)
  dim3 g2(D_ / 128, M_ / 128, 1);
  gemm_wmma_k<0, 0, 3><<<g2, 256, 0, stream>>>(H1H, W2H, b2, x, YF, nullptr,
      M_, D_, DH_, DH_, DH_, D_, D_, 1, 1, Z,Z,Z, Z,Z,Z, Z,Z,Z, 1.0f);

  // ---- layernorm -> d_out
  layernorm_k<<<M_, 256, 0, stream>>>(YF, ln_g, ln_b, (float*)d_out);
}
